// MHCLiteBlock_18769007084525
// MI455X (gfx1250) — matrix operand order repaired and checked by reading, not compiled, not run on hardware
//
#include <hip/hip_runtime.h>
#include <hip/hip_bf16.h>

#define TT  8192
#define NN  4
#define CC  2048
#define NCC 8192

typedef __attribute__((ext_vector_type(16))) __bf16 v16bf;
typedef __attribute__((ext_vector_type(8)))  float  v8f;
typedef __attribute__((ext_vector_type(4)))  __bf16 bf4;
typedef __attribute__((ext_vector_type(4)))  int    v4i;

// pointer-to-16B-vector in global (AS1) / LDS (AS3) address spaces
typedef __attribute__((address_space(1))) v4i* gptr_v4i;
typedef __attribute__((address_space(3))) v4i* lptr_v4i;

// LDS byte-offset of a __shared__ object (generic -> AS3 -> int)
#define LDS_OFF(p) ((unsigned)(size_t)(__attribute__((address_space(3))) void*)(p))

#if __has_builtin(__builtin_amdgcn_tensor_load_to_lds)
#define HAVE_TDM 1
#else
#define HAVE_TDM 0
#endif

#if __has_builtin(__builtin_amdgcn_global_load_async_to_lds_b128)
#define HAVE_ASYNC 1
#else
#define HAVE_ASYNC 0
#endif

__device__ __forceinline__ void wait_asynccnt0() {
#if __has_builtin(__builtin_amdgcn_s_wait_asynccnt)
  __builtin_amdgcn_s_wait_asynccnt(0);
#else
  asm volatile("s_wait_asynccnt 0x0" ::: "memory");
#endif
}

#if HAVE_TDM
typedef unsigned tdm_u4 __attribute__((ext_vector_type(4)));
typedef int      tdm_i8 __attribute__((ext_vector_type(8)));
typedef int      tdm_i4 __attribute__((ext_vector_type(4)));

// TDM: load a 2D bf16 tile (tile_k contiguous elems x tile_rows rows) into LDS.
// Optional LDS padding: after every (2<<pad_interval) DWORDs insert (pad_amount+1) DWORDs.
__device__ __forceinline__ void tdm_load_tile_bf16(
    const __bf16* gsrc, unsigned lds_off,
    unsigned tile_k, unsigned tile_rows, unsigned row_stride_elems,
    unsigned tensor_d0, unsigned tensor_d1,
    unsigned pad_interval, unsigned pad_amount, int pad_en) {
  unsigned long long ga = (unsigned long long)(size_t)gsrc;
  tdm_u4 g0;
  g0[0] = 1u;                                               // count=1, no gather
  g0[1] = lds_off;                                          // lds_addr
  g0[2] = (unsigned)ga;                                     // global_addr[31:0]
  g0[3] = (unsigned)((ga >> 32) & 0x1FFFFFFu) | (2u << 30); // addr[56:32] | type=2
  unsigned w0 = (1u << 16);                                 // data_size = 2 bytes
  if (pad_en)
    w0 |= (1u << 20) | (pad_interval << 22) | (pad_amount << 25);
  tdm_i8 g1;
  g1[0] = (int)w0;                                              // mask/size/pad
  g1[1] = (int)((tensor_d0 & 0xFFFFu) << 16);                   // dim0[15:0]
  g1[2] = (int)(((tensor_d0 >> 16) & 0xFFFFu) |
                ((tensor_d1 & 0xFFFFu) << 16));                 // dim0[31:16]|dim1[15:0]
  g1[3] = (int)(((tensor_d1 >> 16) & 0xFFFFu) |
                ((tile_k & 0xFFFFu) << 16));                    // dim1[31:16]|tile_dim0
  g1[4] = (int)(tile_rows & 0xFFFFu);                           // tile_dim1, tile_dim2=0
  g1[5] = (int)row_stride_elems;                                // dim0_stride[31:0]
  g1[6] = 0;                                                    // stride hi bits
  g1[7] = 0;
  tdm_i4 z4 = (tdm_i4){0, 0, 0, 0};
  tdm_i8 z8 = (tdm_i8){0, 0, 0, 0, 0, 0, 0, 0};
  __builtin_amdgcn_tensor_load_to_lds(g0, g1, z4, z4, z8, 0);
}
#endif

// ---------------------------------------------------------------------------
// K0: convert W_layer (f32, CC x CC) -> bf16
// ---------------------------------------------------------------------------
__global__ __launch_bounds__(256) void k_cvt_wb(const float* __restrict__ w,
                                                __bf16* __restrict__ wb) {
  int i = (blockIdx.x * 256 + threadIdx.x) * 4;
  float4 v = *(const float4*)(w + i);
  bf4 o;
  o[0] = (__bf16)v.x; o[1] = (__bf16)v.y; o[2] = (__bf16)v.z; o[3] = (__bf16)v.w;
  *(bf4*)(wb + i) = o;
}

// ---------------------------------------------------------------------------
// K1: per-row RMS sumsq + raw projection proj[t,0..31] = invrms * (x_row . W_all[j])
// 64 rows per block; W_all chunk staged in LDS (async-to-LDS when available).
// ---------------------------------------------------------------------------
#define KC  256
#define KCP 260
__global__ __launch_bounds__(256) void k_proj(const float* __restrict__ x,
                                              const float* __restrict__ Wall,
                                              float* __restrict__ proj) {
  __shared__ float Wl[32][KCP];
  const int tid  = threadIdx.x;
  const int j    = tid & 31;       // projection column (wave32 lane)
  const int rsub = tid >> 5;       // wave id: row sub-group
  const int rowBase = blockIdx.x * 64;

  float acc[8], sq[8];
#pragma unroll
  for (int rg = 0; rg < 8; ++rg) { acc[rg] = 0.f; sq[rg] = 0.f; }

  for (int kb = 0; kb < NCC; kb += KC) {
    // cooperative stage of W_all[0..31][kb..kb+KC)
#if HAVE_ASYNC
    for (int i = tid; i < 32 * (KC / 4); i += 256) {
      int jr = i / (KC / 4);
      int kk = (i % (KC / 4)) * 4;
      __builtin_amdgcn_global_load_async_to_lds_b128(
          (gptr_v4i)(size_t)(Wall + (size_t)jr * NCC + kb + kk),
          (lptr_v4i)&Wl[jr][kk], 0, 0);
    }
    wait_asynccnt0();
#else
    for (int i = tid; i < 32 * (KC / 4); i += 256) {
      int jr = i / (KC / 4);
      int kk = (i % (KC / 4)) * 4;
      *(float4*)&Wl[jr][kk] = *(const float4*)(Wall + (size_t)jr * NCC + kb + kk);
    }
#endif
    __syncthreads();

#pragma unroll 1
    for (int rg = 0; rg < 8; ++rg) {
      const int row = rowBase + rsub * 8 + rg;
      const float4* xv = (const float4*)(x + (size_t)row * NCC + kb);
      float s = 0.f, q = 0.f;
#pragma unroll 4
      for (int kk = 0; kk < KC / 4; ++kk) {
        float4 xx = xv[kk];                       // lane-uniform (broadcast)
        float4 ww = *(const float4*)&Wl[j][kk * 4];
        s += xx.x * ww.x + xx.y * ww.y + xx.z * ww.z + xx.w * ww.w;
        if (j == 0)  // lane0 owns sum-of-squares (sees the full row)
          q += xx.x * xx.x + xx.y * xx.y + xx.z * xx.z + xx.w * xx.w;
      }
      acc[rg] += s; sq[rg] += q;
    }
    __syncthreads();
  }

#pragma unroll
  for (int rg = 0; rg < 8; ++rg) {
    const int row = rowBase + rsub * 8 + rg;
    float ss  = __shfl(sq[rg], 0, 32);            // wave32 broadcast
    float inv = rsqrtf(ss * (1.0f / NCC) + 1.1920929e-7f);
    proj[row * 32 + j] = acc[rg] * inv;
  }
}

// ---------------------------------------------------------------------------
// K2: gates: h_pre, h_post, softmax over 24 factors, H_merged = a@perm - outer
// ---------------------------------------------------------------------------
__global__ __launch_bounds__(256) void k_gates(const float* __restrict__ proj,
                                               const float* __restrict__ b_all,
                                               const float* __restrict__ a_pre_p,
                                               const float* __restrict__ a_post_p,
                                               const float* __restrict__ a_res_p,
                                               const float* __restrict__ perm,
                                               float* __restrict__ hpre,
                                               float* __restrict__ hpost,
                                               float* __restrict__ Hm) {
  const int t = blockIdx.x * 256 + threadIdx.x;
  const float* p = proj + t * 32;
  const float a_pre = a_pre_p[0], a_post = a_post_p[0], a_res = a_res_p[0];

  float hp[4], ho[4];
#pragma unroll
  for (int n = 0; n < 4; ++n)
    hp[n] = 1.0f / (1.0f + expf(-(a_pre * p[n] + b_all[n])));
#pragma unroll
  for (int n = 0; n < 4; ++n)
    ho[n] = 2.0f / (1.0f + expf(-(a_post * p[4 + n] + b_all[4 + n])));

  float z[24], zmax = -3.4e38f;
#pragma unroll
  for (int q = 0; q < 24; ++q) {
    z[q] = a_res * p[8 + q] + b_all[8 + q];
    zmax = fmaxf(zmax, z[q]);
  }
  float zs = 0.f;
#pragma unroll
  for (int q = 0; q < 24; ++q) { z[q] = expf(z[q] - zmax); zs += z[q]; }
  const float rs = 1.0f / zs;

  float Hr[16];
#pragma unroll
  for (int i = 0; i < 16; ++i) Hr[i] = 0.f;
#pragma unroll
  for (int q = 0; q < 24; ++q) {
    float a = z[q] * rs;
#pragma unroll
    for (int i = 0; i < 16; ++i) Hr[i] += a * perm[q * 16 + i];
  }

#pragma unroll
  for (int n = 0; n < 4; ++n) {
    hpre[t * 4 + n]  = hp[n];
    hpost[t * 4 + n] = ho[n];
#pragma unroll
    for (int m = 0; m < 4; ++m)
      Hm[t * 16 + n * 4 + m] = Hr[n * 4 + m] - ho[n] * hp[m];
  }
}

// ---------------------------------------------------------------------------
// K3: layer_input[t,c] = sum_n h_pre[t,n] * x[t,n,c]  -> bf16 (GEMM A operand)
// ---------------------------------------------------------------------------
__global__ __launch_bounds__(256) void k_li(const float* __restrict__ x,
                                            const float* __restrict__ hpre,
                                            __bf16* __restrict__ li) {
  const int idx = blockIdx.x * 256 + threadIdx.x;
  const int t = idx >> 9, c4 = (idx & 511) * 4;
  const float* hp = hpre + t * 4;
  float ax = 0.f, ay = 0.f, az = 0.f, aw = 0.f;
#pragma unroll
  for (int n = 0; n < 4; ++n) {
    float4 xx = *(const float4*)(x + ((size_t)(t * 4 + n)) * CC + c4);
    float h = hp[n];
    ax += h * xx.x; ay += h * xx.y; az += h * xx.z; aw += h * xx.w;
  }
  bf4 o; o[0] = (__bf16)ax; o[1] = (__bf16)ay; o[2] = (__bf16)az; o[3] = (__bf16)aw;
  *(bf4*)(li + (size_t)t * CC + c4) = o;
}

// ---------------------------------------------------------------------------
// K4: layer_output[T,CC] = li(bf16) @ Wb(bf16, row = out col, K inner).T + bias
// Block tile 128x128, 8 waves (2 M x 4 N), wave tile 64x32 = 4x2 WMMA tiles.
// A/B tiles staged by the Tensor Data Mover when available (with LDS padding
// 16 DWORDs data + 4 DWORDs pad => 80B rows matching the [128][40] layout).
// ---------------------------------------------------------------------------
__global__ __launch_bounds__(256) void k_gemm(const __bf16* __restrict__ A,
                                              const __bf16* __restrict__ B,
                                              const float* __restrict__ bias,
                                              float* __restrict__ Cout) {
  __shared__ __bf16 As[128][40];   // pad 32->40: conflict-free frag reads, 16B-aligned rows
  __shared__ __bf16 Bs[128][40];

  const int tid  = threadIdx.x;
  const int lane = tid & 31;
  const int w    = tid >> 5;
  const int wm   = w >> 2;                // 0..1
  const int wn   = w & 3;                 // 0..3
  const int m0   = blockIdx.y * 128;
  const int n0   = blockIdx.x * 128;
  const int lr   = tid >> 1;              // staging row 0..127
  const int lh   = (tid & 1) * 16;        // staging half (16 bf16 = 32B)
  (void)lr; (void)lh;

  v8f c[4][2];
#pragma unroll
  for (int mi = 0; mi < 4; ++mi)
#pragma unroll
    for (int ni = 0; ni < 2; ++ni) c[mi][ni] = (v8f){};

  union Frag { v16bf v; unsigned u[8]; };

  for (int k0 = 0; k0 < CC; k0 += 32) {
#if HAVE_TDM
    // TDM staging: wave0 -> A tile, wave1 -> B tile; pad_interval=16 DW (code 3),
    // pad_amount=4 DW (code 3) => 80-byte LDS rows.
    if (w == 0)
      tdm_load_tile_bf16(A + (size_t)m0 * CC + k0, LDS_OFF(&As[0][0]),
                         32, 128, CC, CC, TT, 3, 3, 1);
    if (w == 1)
      tdm_load_tile_bf16(B + (size_t)n0 * CC + k0, LDS_OFF(&Bs[0][0]),
                         32, 128, CC, CC, CC, 3, 3, 1);
    if (w < 2)
      __builtin_amdgcn_s_wait_tensorcnt(0);
#else
    { // cooperative stage: A rows m0..m0+127, B rows (out cols) n0..n0+127; K slab 32
      const uint4* sa = (const uint4*)(A + (size_t)(m0 + lr) * CC + k0 + lh);
      uint4* da = (uint4*)&As[lr][lh];
      da[0] = sa[0]; da[1] = sa[1];
      const uint4* sb = (const uint4*)(B + (size_t)(n0 + lr) * CC + k0 + lh);
      uint4* db = (uint4*)&Bs[lr][lh];
      db[0] = sb[0]; db[1] = sb[1];
    }
    if (k0 + 32 < CC) {  // global_prefetch_b8 of next K slab
      __builtin_prefetch(A + (size_t)(m0 + lr) * CC + k0 + 32 + lh, 0, 1);
      __builtin_prefetch(B + (size_t)(n0 + lr) * CC + k0 + 32 + lh, 0, 1);
    }
#endif
    __syncthreads();

    // B fragments: this code's contiguous k order: lanes0-15 = col, K0..15 in words0..7; lanes16-31 K16..31
    Frag bf[2];
    const int kb2 = (lane >> 4) << 3;
#pragma unroll
    for (int ni = 0; ni < 2; ++ni) {
      const int col = wn * 32 + ni * 16 + (lane & 15);
      const unsigned* Br = (const unsigned*)&Bs[col][0];
#pragma unroll
      for (int v = 0; v < 8; ++v) bf[ni].u[v] = Br[kb2 + v];
    }

    // A fragments: this code's contiguous k order (words 0..7 / 8..15, matching B)
    const int kb = (lane >> 4) << 3;
#pragma unroll
    for (int mi = 0; mi < 4; ++mi) {
      Frag af;
      const int row = wm * 64 + mi * 16 + (lane & 15);
      const unsigned* Ar = (const unsigned*)&As[row][0];
#pragma unroll
      for (int v = 0; v < 8; ++v)
        af.u[v] = Ar[v + kb];
#pragma unroll
      for (int ni = 0; ni < 2; ++ni)
        c[mi][ni] = __builtin_amdgcn_wmma_f32_16x16x32_bf16(
            false, af.v, false, bf[ni].v, (short)0, c[mi][ni], false, false);
    }
    __syncthreads();
  }

  // epilogue: C/D layout: VGPR r -> M = r + 8*(lane>=16), N = lane&15
#pragma unroll
  for (int mi = 0; mi < 4; ++mi)
#pragma unroll
    for (int ni = 0; ni < 2; ++ni) {
      const int col = n0 + wn * 32 + ni * 16 + (lane & 15);
      const float bv = bias[col];
      const int mb = m0 + wm * 64 + mi * 16 + ((lane >> 4) << 3);
#pragma unroll
      for (int r = 0; r < 8; ++r)
        Cout[(size_t)(mb + r) * CC + col] = c[mi][ni][r] + bv;
    }
}

// ---------------------------------------------------------------------------
// K5: out[t,n,c] = sum_m Hm[t,n,m]*x[t,m,c] + h_post[t,n]*layer_output[t,c]
// ---------------------------------------------------------------------------
__global__ __launch_bounds__(256) void k_mix(const float* __restrict__ x,
                                             const float* __restrict__ Hm,
                                             const float* __restrict__ hpost,
                                             const float* __restrict__ lo,
                                             float* __restrict__ out) {
  const int idx = blockIdx.x * 256 + threadIdx.x;
  const int t = idx >> 9, c4 = (idx & 511) * 4;

  float hm[16], hp[4];
#pragma unroll
  for (int i = 0; i < 16; ++i) hm[i] = Hm[t * 16 + i];
#pragma unroll
  for (int n = 0; n < 4; ++n) hp[n] = hpost[t * 4 + n];

  float4 xs[4];
#pragma unroll
  for (int m = 0; m < 4; ++m)
    xs[m] = *(const float4*)(x + ((size_t)(t * 4 + m)) * CC + c4);
  float4 lv = *(const float4*)(lo + (size_t)t * CC + c4);

#pragma unroll
  for (int n = 0; n < 4; ++n) {
    float4 o;
    o.x = hp[n] * lv.x; o.y = hp[n] * lv.y; o.z = hp[n] * lv.z; o.w = hp[n] * lv.w;
#pragma unroll
    for (int m = 0; m < 4; ++m) {
      const float h = hm[n * 4 + m];
      o.x += h * xs[m].x; o.y += h * xs[m].y; o.z += h * xs[m].z; o.w += h * xs[m].w;
    }
    *(float4*)(out + ((size_t)(t * 4 + n)) * CC + c4) = o;
  }
}

// ---------------------------------------------------------------------------
extern "C" void kernel_launch(void* const* d_in, const int* in_sizes, int n_in,
                              void* d_out, int out_size, void* d_ws, size_t ws_size,
                              hipStream_t stream) {
  const float* x       = (const float*)d_in[0];
  const float* W_all   = (const float*)d_in[1];
  const float* b_all   = (const float*)d_in[2];
  const float* a_pre   = (const float*)d_in[3];
  const float* a_post  = (const float*)d_in[4];
  const float* a_res   = (const float*)d_in[5];
  const float* perm    = (const float*)d_in[6];
  const float* W_layer = (const float*)d_in[7];
  const float* b_layer = (const float*)d_in[8];
  float* out = (float*)d_out;

  // workspace layout (256B aligned)
  char* p = (char*)d_ws;
  auto take = [&](size_t bytes) {
    char* r = p;
    p += (bytes + 255) & ~(size_t)255;
    return r;
  };
  float*  proj  = (float*)take((size_t)TT * 32 * 4);
  float*  hpre  = (float*)take((size_t)TT * 4 * 4);
  float*  hpost = (float*)take((size_t)TT * 4 * 4);
  float*  Hm    = (float*)take((size_t)TT * 16 * 4);
  __bf16* li    = (__bf16*)take((size_t)TT * CC * 2);
  __bf16* Wb    = (__bf16*)take((size_t)CC * CC * 2);
  float*  lo    = (float*)take((size_t)TT * CC * 4);
  (void)in_sizes; (void)n_in; (void)out_size; (void)ws_size;

  k_cvt_wb<<<(CC * CC) / 1024, 256, 0, stream>>>(W_layer, Wb);
  k_proj<<<TT / 64, 256, 0, stream>>>(x, W_all, proj);
  k_gates<<<TT / 256, 256, 0, stream>>>(proj, b_all, a_pre, a_post, a_res, perm,
                                        hpre, hpost, Hm);
  k_li<<<(TT * (CC / 4)) / 256, 256, 0, stream>>>(x, hpre, li);
  k_gemm<<<dim3(CC / 128, TT / 128), 256, 0, stream>>>(li, Wb, b_layer, lo);
  k_mix<<<(TT * (CC / 4)) / 256, 256, 0, stream>>>(x, Hm, hpost, lo, out);
}
